// WINNERHierarchicalDecomposer_34454227648861
// MI455X (gfx1250) — matrix operand order repaired and checked by reading, not compiled, not run on hardware
//
#include <hip/hip_runtime.h>
#include <hip/hip_bf16.h>
#include <math.h>

typedef _Float16 half_t;
typedef __attribute__((ext_vector_type(16))) _Float16 v16h;
typedef __attribute__((ext_vector_type(8)))  _Float16 v8h;
typedef __attribute__((ext_vector_type(4)))  _Float16 v4h;
typedef __attribute__((ext_vector_type(8)))  float    v8f;
typedef __attribute__((ext_vector_type(4)))  int      v4i;

#define HSZ 768
#define SEQ 1000
#define SEQP 1024
#define BATCH 4
#define NHEAD 12
#define HDIM 64
#define NLVL 3

// ---- optional CDNA5 async global->LDS copy (guarded; fallback = vector ld/st)
#if defined(__has_builtin)
#if __has_builtin(__builtin_amdgcn_global_load_async_to_lds_b128)
#define HAVE_ASYNC 1
#endif
#endif
#ifndef HAVE_ASYNC
#define HAVE_ASYNC 0
#endif

#if HAVE_ASYNC
#define AS1 __attribute__((address_space(1)))
#define AS3 __attribute__((address_space(3)))
__device__ __forceinline__ void async_cp16(const half_t* g, half_t* l) {
    // builtin signature (from hipcc diagnostic): (v4i addrspace(1)*, v4i addrspace(3)*, imm offset, imm cpol)
    __builtin_amdgcn_global_load_async_to_lds_b128((AS1 v4i*)(half_t*)g, (AS3 v4i*)l, 0, 0);
}
#endif

// ---------------------------------------------------------------------------
// Tiled f16-WMMA GEMM:  out = A[M,K](f16) @ W[K,N](f16) + bias
// Block tile 128x64, K-step 32, 8 waves; A tile async-copied to LDS (f16,
// no conversion in hot loop), W tile staged transposed. Epilogue writes
// f32 (D) and/or scaled f16 (Dh).
// ---------------------------------------------------------------------------
__global__ __launch_bounds__(256)
void gemm_f16wmma_kernel(const half_t* __restrict__ A, const half_t* __restrict__ W,
                         const float* __restrict__ bias, const float* __restrict__ res,
                         float* __restrict__ D, half_t* __restrict__ Dh, float hscale,
                         int M, int N, int K)
{
    __shared__ half_t lA[128 * 32];   // [row][k]
    __shared__ half_t lBt[64 * 32];   // transposed: [n][k]
    const int tid  = threadIdx.x;
    const int wid  = tid >> 5;
    const int lane = tid & 31;
    const int h    = lane >> 4;
    const int ln16 = lane & 15;
    const int m0 = blockIdx.x * 128;
    const int n0 = blockIdx.y * 64;

    v8f acc[4] = {};

    for (int kt = 0; kt < K; kt += 32) {
        // --- A tile 128x32 halves: 512 x 16B chunks, 2 per thread ---
#pragma unroll
        for (int cc = 0; cc < 2; ++cc) {
            const int ch  = tid + cc * 256;
            const int row = ch >> 2;
            const int seg = (ch & 3) * 8;
            const int gm  = m0 + row;
            if (gm < M) {
                const half_t* src = A + (size_t)gm * K + kt + seg;
#if HAVE_ASYNC
                async_cp16(src, &lA[row * 32 + seg]);
#else
                *(v8h*)&lA[row * 32 + seg] = *(const v8h*)src;
#endif
            }
        }
        // --- W tile 32x64 halves staged transposed into lBt[n][k] ---
        {
            const int k    = tid >> 3;
            const int nseg = (tid & 7) * 8;
            const v8h w = *(const v8h*)(W + (size_t)(kt + k) * N + n0 + nseg);
#pragma unroll
            for (int i = 0; i < 8; ++i) lBt[(nseg + i) * 32 + k] = w[i];
            if (kt + 32 < K)
                __builtin_prefetch(W + (size_t)(kt + 32 + k) * N + n0 + nseg, 0, 1);
        }
#if HAVE_ASYNC
        asm volatile("s_wait_asynccnt 0" ::: "memory");
#endif
        __syncthreads();

        // --- fragments (CDNA5 16-bit WMMA lane layouts) ---
        v16h a;
        const half_t* arow = &lA[(wid * 16 + ln16) * 32];
#pragma unroll
        for (int e = 0; e < 8; ++e)  a[e] = arow[e + 16 * h];
#pragma unroll
        for (int e = 8; e < 16; ++e) a[e] = arow[e + 16 * h];

#pragma unroll
        for (int sub = 0; sub < 4; ++sub) {
            v16h b;
            const half_t* bcol = &lBt[(sub * 16 + ln16) * 32 + 16 * h];
#pragma unroll
            for (int e = 0; e < 16; ++e) b[e] = bcol[e];
            acc[sub] = __builtin_amdgcn_wmma_f32_16x16x32_f16(
                false, a, false, b, (short)0, acc[sub], false, false);
        }
        __syncthreads();
    }

    // --- epilogue: C layout row = j + 8h, col = ln16 ---
#pragma unroll
    for (int sub = 0; sub < 4; ++sub) {
        const int gn = n0 + sub * 16 + ln16;
#pragma unroll
        for (int j = 0; j < 8; ++j) {
            const int gm = m0 + wid * 16 + j + 8 * h;
            if (gm < M) {
                float v = acc[sub][j] + bias[gn];
                if (res) v += res[(size_t)gm * N + gn];
                if (D)  D[(size_t)gm * N + gn]  = v;
                if (Dh) Dh[(size_t)gm * N + gn] = (half_t)(v * hscale);
            }
        }
    }
}

// ---------------------------------------------------------------------------
// LayerNorm rows (H=768): out = [relu]( (x-m)*rstd*g + be ) [+post]; f32 + f16
// ---------------------------------------------------------------------------
__global__ __launch_bounds__(256)
void ln_act_kernel(const float* __restrict__ in, const float* __restrict__ g,
                   const float* __restrict__ be, const float* __restrict__ post,
                   float* __restrict__ out, half_t* __restrict__ outh, int relu)
{
    __shared__ float red[256];
    const int row = blockIdx.x;
    const int t = threadIdx.x;
    const float* x = in + (size_t)row * HSZ;
    float v0 = x[t], v1 = x[t + 256], v2 = x[t + 512];
    red[t] = v0 + v1 + v2;
    __syncthreads();
    for (int o = 128; o > 0; o >>= 1) { if (t < o) red[t] += red[t + o]; __syncthreads(); }
    const float mean = red[0] * (1.0f / 768.0f);
    __syncthreads();
    const float d0 = v0 - mean, d1 = v1 - mean, d2 = v2 - mean;
    red[t] = d0 * d0 + d1 * d1 + d2 * d2;
    __syncthreads();
    for (int o = 128; o > 0; o >>= 1) { if (t < o) red[t] += red[t + o]; __syncthreads(); }
    const float rstd = rsqrtf(red[0] * (1.0f / 768.0f) + 1e-5f);

    const float dd[3] = { d0, d1, d2 };
#pragma unroll
    for (int k = 0; k < 3; ++k) {
        const int c = t + k * 256;
        float y = dd[k] * rstd * g[c] + be[c];
        if (relu) y = fmaxf(y, 0.f);
        if (post) y += post[(size_t)row * HSZ + c];
        out[(size_t)row * HSZ + c] = y;
        if (outh) outh[(size_t)row * HSZ + c] = (half_t)y;
    }
}

// inp(f32) = cur + level_emb ; fwp(f16) = inp + pos*scale
__global__ void add_pos_kernel(const float* __restrict__ cur, const float* __restrict__ lev,
                               const float* __restrict__ pos, float scale,
                               float* __restrict__ inp, half_t* __restrict__ fwph)
{
    const int i = blockIdx.x * 256 + threadIdx.x;     // B*S*H exact
    const int c = i % HSZ;
    const int s = (i / HSZ) % SEQ;
    const float a = cur[i] + lev[c];
    inp[i] = a;
    fwph[i] = (half_t)(a + pos[(size_t)s * HSZ + c] * scale);
}

// f32 -> f16 bulk convert (n multiple of 4)
__global__ void cvt_f16_kernel(const float* __restrict__ s, half_t* __restrict__ d, int n4)
{
    const int i = blockIdx.x * 256 + threadIdx.x;
    if (i < n4) {
        const float4 v = ((const float4*)s)[i];
        v4h hv = { (half_t)v.x, (half_t)v.y, (half_t)v.z, (half_t)v.w };
        *(v4h*)(d + (size_t)i * 4) = hv;
    }
}

// V[b,s,h,d] (f16 row-major) -> Vt[b,h,d,s] padded S->1024, zero-filled tail
__global__ void transposeV_kernel(const half_t* __restrict__ Vh, half_t* __restrict__ Vt)
{
    const int i = blockIdx.x * 256 + threadIdx.x;     // B*NHEAD*HDIM*SEQP exact
    const int s = i & (SEQP - 1);
    const int rest = i >> 10;
    const int d = rest & (HDIM - 1);
    const int bh = rest >> 6;
    const int hh = bh % NHEAD, b = bh / NHEAD;
    half_t v = (half_t)0.f;
    if (s < SEQ) v = Vh[(size_t)(b * SEQ + s) * HSZ + hh * HDIM + d];
    Vt[i] = v;
}

// ---------------------------------------------------------------------------
// Banded attention (levels 0/1): one wave per (q, head, b); <=11 keys.
// Q already pre-scaled by 1/sqrt(HD) in the projection epilogue.
// ---------------------------------------------------------------------------
__global__ __launch_bounds__(32)
void band_attn_kernel(const half_t* __restrict__ Q, const half_t* __restrict__ K,
                      const half_t* __restrict__ V, half_t* __restrict__ O, int win)
{
    const int q = blockIdx.x, head = blockIdx.y, b = blockIdx.z;
    const int lane = threadIdx.x;
    const half_t* qp = Q + ((size_t)(b * SEQ + q) * HSZ + head * HDIM);
    const float q0 = (float)qp[lane], q1 = (float)qp[lane + 32];
    int k0 = q - win; if (k0 < 0) k0 = 0;
    int k1 = q + win; if (k1 > SEQ - 1) k1 = SEQ - 1;
    const int nk = k1 - k0 + 1;
    float sc[16];
    float mx = -1e30f;
    for (int i = 0; i < nk; ++i) {
        const half_t* kp = K + ((size_t)(b * SEQ + k0 + i) * HSZ + head * HDIM);
        float p = q0 * (float)kp[lane] + q1 * (float)kp[lane + 32];
        for (int m = 1; m < 32; m <<= 1) p += __shfl_xor(p, m, 32);
        sc[i] = p; mx = fmaxf(mx, p);
    }
    float sum = 0.f;
    for (int i = 0; i < nk; ++i) { sc[i] = __expf(sc[i] - mx); sum += sc[i]; }
    const float inv = 1.f / sum;
    float o0 = 0.f, o1 = 0.f;
    for (int i = 0; i < nk; ++i) {
        const half_t* vp = V + ((size_t)(b * SEQ + k0 + i) * HSZ + head * HDIM);
        o0 += sc[i] * (float)vp[lane]; o1 += sc[i] * (float)vp[lane + 32];
    }
    half_t* op = O + ((size_t)(b * SEQ + q) * HSZ + head * HDIM);
    op[lane] = (half_t)(o0 * inv); op[lane + 32] = (half_t)(o1 * inv);
}

// ---------------------------------------------------------------------------
// Dense flash attention (level 2), WMMA, all-f16 operands, contiguous frag
// loads (V pre-transposed). Online softmax; P relaid C->A layout via LDS.
// ---------------------------------------------------------------------------
__global__ __launch_bounds__(32)
void flash_attn_kernel(const half_t* __restrict__ Q, const half_t* __restrict__ K,
                       const half_t* __restrict__ Vt, half_t* __restrict__ O)
{
    __shared__ half_t pbuf[16 * 32];
    const int qt = blockIdx.x, head = blockIdx.y, b = blockIdx.z;
    const int lane = threadIdx.x;
    const int h = lane >> 4, ln16 = lane & 15;
    const int q0 = qt * 16;

    // Q fragments (pre-scaled); clamp tail rows (their outputs are discarded)
    v16h aq0, aq1;
    {
        int s = q0 + ln16; if (s >= SEQ) s = SEQ - 1;
        const half_t* qp = Q + ((size_t)(b * SEQ + s) * HSZ + head * HDIM);
        const v8h c0 = *(const v8h*)(qp + 16 * h);
        const v8h c1 = *(const v8h*)(qp + 16 * h + 8);
        const v8h d0 = *(const v8h*)(qp + 32 + 16 * h);
        const v8h d1 = *(const v8h*)(qp + 32 + 16 * h + 8);
#pragma unroll
        for (int e = 0; e < 8; ++e) {
            aq0[e] = c0[e]; aq0[e + 8] = c1[e];
            aq1[e] = d0[e]; aq1[e + 8] = d1[e];
        }
    }

    v8f o[4] = {};
    float mrun[8], lrun[8];
#pragma unroll
    for (int j = 0; j < 8; ++j) { mrun[j] = -1e30f; lrun[j] = 0.f; }

    for (int kb = 0; kb < SEQ; kb += 32) {
        // ---- score tiles: two 16x16 key tiles ----
        v8f sc[2];
#pragma unroll
        for (int kt = 0; kt < 2; ++kt) {
            const int key = kb + kt * 16 + ln16;
            const int kr = (key < SEQ) ? key : SEQ - 1;
            const half_t* kp = K + ((size_t)(b * SEQ + kr) * HSZ + head * HDIM);
            const v8h k0a = *(const v8h*)(kp + 16 * h);
            const v8h k0b = *(const v8h*)(kp + 16 * h + 8);
            const v8h k1a = *(const v8h*)(kp + 32 + 16 * h);
            const v8h k1b = *(const v8h*)(kp + 32 + 16 * h + 8);
            v16h bk0, bk1;
#pragma unroll
            for (int e = 0; e < 8; ++e) {
                bk0[e] = k0a[e]; bk0[e + 8] = k0b[e];
                bk1[e] = k1a[e]; bk1[e + 8] = k1b[e];
            }
            v8f z = {};
            z = __builtin_amdgcn_wmma_f32_16x16x32_f16(false, aq0, false, bk0, (short)0, z, false, false);
            z = __builtin_amdgcn_wmma_f32_16x16x32_f16(false, aq1, false, bk1, (short)0, z, false, false);
            if (key >= SEQ) {
#pragma unroll
                for (int j = 0; j < 8; ++j) z[j] = -1e9f;
            }
            sc[kt] = z;
        }
        // ---- online softmax per row (row = j + 8h; width-16 reductions) ----
        float pv0[8], pv1[8];
#pragma unroll
        for (int j = 0; j < 8; ++j) {
            float t = fmaxf(sc[0][j], sc[1][j]);
            for (int m = 1; m < 16; m <<= 1) t = fmaxf(t, __shfl_xor(t, m, 16));
            const float mnew = fmaxf(mrun[j], t);
            const float corr = __expf(mrun[j] - mnew);
            const float p0 = __expf(sc[0][j] - mnew);
            const float p1 = __expf(sc[1][j] - mnew);
            float r = p0 + p1;
            for (int m = 1; m < 16; m <<= 1) r += __shfl_xor(r, m, 16);
            lrun[j] = lrun[j] * corr + r;
            mrun[j] = mnew;
#pragma unroll
            for (int sub = 0; sub < 4; ++sub) o[sub][j] *= corr;
            pv0[j] = p0; pv1[j] = p1;
        }
        // ---- P: C-layout -> A-layout through LDS ----
#pragma unroll
        for (int j = 0; j < 8; ++j) {
            pbuf[(j + 8 * h) * 32 + ln16]      = (half_t)pv0[j];
            pbuf[(j + 8 * h) * 32 + 16 + ln16] = (half_t)pv1[j];
        }
        asm volatile("s_wait_dscnt 0" ::: "memory");
        v16h ap;
#pragma unroll
        for (int e = 0; e < 16; ++e) {
            const int k = e + 16 * h;
            ap[e] = pbuf[ln16 * 32 + k];
        }
        // ---- ctx += P(16x32) @ V(32x64), V transposed -> contiguous loads ----
#pragma unroll
        for (int sub = 0; sub < 4; ++sub) {
            const half_t* vp = Vt + (((size_t)(b * NHEAD + head) * HDIM + sub * 16 + ln16) << 10)
                                  + kb + 16 * h;
            const v8h va = *(const v8h*)vp;
            const v8h vb = *(const v8h*)(vp + 8);
            v16h bv;
#pragma unroll
            for (int e = 0; e < 8; ++e) { bv[e] = va[e]; bv[e + 8] = vb[e]; }
            o[sub] = __builtin_amdgcn_wmma_f32_16x16x32_f16(false, ap, false, bv, (short)0, o[sub], false, false);
        }
    }
    // ---- finalize ----
#pragma unroll
    for (int sub = 0; sub < 4; ++sub) {
#pragma unroll
        for (int j = 0; j < 8; ++j) {
            const int s = q0 + j + 8 * h;
            if (s < SEQ)
                O[(size_t)(b * SEQ + s) * HSZ + head * HDIM + sub * 16 + ln16] =
                    (half_t)(o[sub][j] / lrun[j]);
        }
    }
}

// ---------------------------------------------------------------------------
// Small tail kernels (f32)
// ---------------------------------------------------------------------------
__global__ void meanS_kernel(const float* __restrict__ f, float* __restrict__ outp)
{
    const int i = blockIdx.x * 256 + threadIdx.x;   // B*H exact
    const int b = i / HSZ, c = i % HSZ;
    const float* p = f + (size_t)b * SEQ * HSZ + c;
    float s = 0.f;
    for (int t = 0; t < SEQ; ++t) s += p[(size_t)t * HSZ];
    outp[(size_t)b * (3 * HSZ) + c] = s * (1.0f / SEQ);
}

__global__ void sp1_kernel(const float* __restrict__ cm, const float* __restrict__ W1,
                           const float* __restrict__ b1, float* __restrict__ hid)
{
    const int i = blockIdx.x * 256 + threadIdx.x;
    if (i >= BATCH * HSZ) return;
    const int b = i / HSZ, j = i % HSZ;
    const float* c = cm + (size_t)b * 3 * HSZ;
    float s = b1[j];
    for (int k = 0; k < 3 * HSZ; ++k) s += c[k] * W1[(size_t)k * HSZ + j];
    hid[i] = fmaxf(s, 0.f);
}

__global__ void sp2_kernel(const float* __restrict__ hid, const float* __restrict__ W2,
                           const float* __restrict__ b2, float* __restrict__ rel)
{
    const int i = threadIdx.x;
    if (i >= BATCH * NLVL) return;
    const int b = i / NLVL, l = i % NLVL;
    float s = b2[l];
    for (int j = 0; j < HSZ; ++j) s += hid[b * HSZ + j] * W2[j * NLVL + l];
    rel[i] = 1.f / (1.f + __expf(-s));
}

__global__ void gate_concat_kernel(const float* __restrict__ f, const float* __restrict__ rel,
                                   half_t* __restrict__ cc, int l)
{
    const int i = blockIdx.x * 256 + threadIdx.x;   // B*S*H exact
    const int b = i / (SEQ * HSZ);
    cc[(size_t)(i / HSZ) * (3 * HSZ) + l * HSZ + (i % HSZ)] = (half_t)(f[i] * rel[b * NLVL + l]);
}

__global__ void node_kernel(const float* __restrict__ cm, const float* __restrict__ rel,
                            const float* __restrict__ nW, const float* __restrict__ nb,
                            float* __restrict__ node)
{
    const int i = blockIdx.x * 256 + threadIdx.x;
    if (i >= BATCH * HSZ) return;
    const int b = i / HSZ, g = i % HSZ;
    const float r2 = rel[b * NLVL + 2];
    const float* base = cm + (size_t)b * 3 * HSZ + 2 * HSZ;
    float s = nb[g];
    for (int c = 0; c < HSZ; ++c) s += (r2 * base[c]) * nW[(size_t)c * HSZ + g];
    node[i] = fmaxf(s, 0.f);
}

__global__ void edge1_kernel(const float* __restrict__ node, const float* __restrict__ attr,
                             const float* __restrict__ eW1, const float* __restrict__ eb1,
                             float* __restrict__ ehid)
{
    const int i = blockIdx.x * 256 + threadIdx.x;
    if (i >= BATCH * 3 * HSZ) return;
    const int g = i % HSZ, p = (i / HSZ) % 3, b = i / (3 * HSZ);
    const int pi[3] = {0, 0, 1}, pj[3] = {1, 2, 2};
    const float* nd = node + (size_t)b * HSZ;
    const float* ai = attr + (size_t)pi[p] * HSZ;
    const float* aj = attr + (size_t)pj[p] * HSZ;
    float s = eb1[g];
    for (int c = 0; c < HSZ; ++c) s += (nd[c] + ai[c]) * eW1[(size_t)c * HSZ + g];
    for (int c = 0; c < HSZ; ++c) s += (nd[c] + aj[c]) * eW1[(size_t)(HSZ + c) * HSZ + g];
    ehid[i] = fmaxf(s, 0.f);
}

__global__ void edge2_kernel(const float* __restrict__ ehid, const float* __restrict__ eW2,
                             const float* __restrict__ eb2, float* __restrict__ edges)
{
    const int i = threadIdx.x;
    if (i >= BATCH * 3) return;
    float s = eb2[0];
    for (int g = 0; g < HSZ; ++g) s += ehid[(size_t)i * HSZ + g] * eW2[g];
    edges[i] = 1.f / (1.f + __expf(-s));
}

__global__ void graph_kernel(const float* __restrict__ node, const float* __restrict__ attr,
                             const float* __restrict__ edges, float* __restrict__ graph)
{
    const int i = blockIdx.x * 256 + threadIdx.x;
    if (i >= BATCH * HSZ) return;
    const int b = i / HSZ, c = i % HSZ;
    const float amean = (attr[c] + attr[HSZ + c] + attr[2 * HSZ + c]) * (1.f / 3.f);
    const float nrep = node[i] + amean;
    const float emean = (edges[b * 3] + edges[b * 3 + 1] + edges[b * 3 + 2]) * (1.f / 3.f);
    graph[i] = nrep * (1.f + emean);
}

__global__ void final_add_kernel(const float* __restrict__ fused, const float* __restrict__ graph,
                                 float* __restrict__ out)
{
    const int i = blockIdx.x * 256 + threadIdx.x;   // B*S*H exact
    const int b = i / (SEQ * HSZ), c = i % HSZ;
    out[i] = fused[i] + graph[b * HSZ + c];
}

// ---------------------------------------------------------------------------
extern "C" void kernel_launch(void* const* d_in, const int* in_sizes, int n_in,
                              void* d_out, int out_size, void* d_ws, size_t ws_size,
                              hipStream_t stream)
{
    const float* x       = (const float*)d_in[0];
    const float* sem_W   = (const float*)d_in[1];
    const float* sem_b   = (const float*)d_in[2];
    const float* sem_g   = (const float*)d_in[3];
    const float* sem_be  = (const float*)d_in[4];
    const float* pos_emb = (const float*)d_in[5];
    const float* Wq = (const float*)d_in[6];  const float* bq = (const float*)d_in[7];
    const float* Wk = (const float*)d_in[8];  const float* bk = (const float*)d_in[9];
    const float* Wv = (const float*)d_in[10]; const float* bv = (const float*)d_in[11];
    const float* Wo = (const float*)d_in[12]; const float* bo = (const float*)d_in[13];
    const float* ln_g = (const float*)d_in[14]; const float* ln_b = (const float*)d_in[15];
    const float* level_emb = (const float*)d_in[16];
    const float* sp_W1 = (const float*)d_in[17]; const float* sp_b1 = (const float*)d_in[18];
    const float* sp_W2 = (const float*)d_in[19]; const float* sp_b2 = (const float*)d_in[20];
    const float* node_W = (const float*)d_in[21]; const float* node_b = (const float*)d_in[22];
    const float* edge_W1 = (const float*)d_in[23]; const float* edge_b1 = (const float*)d_in[24];
    const float* edge_W2 = (const float*)d_in[25]; const float* edge_b2 = (const float*)d_in[26];
    const float* attr_emb = (const float*)d_in[27];
    const float* fus_W = (const float*)d_in[28]; const float* fus_b = (const float*)d_in[29];
    const float* fus_g = (const float*)d_in[30]; const float* fus_be = (const float*)d_in[31];

    const int M = BATCH * SEQ;                 // 4000
    const size_t E  = (size_t)M * HSZ;         // 3,072,000
    const size_t HH = (size_t)HSZ * HSZ;
    float* ws = (float*)d_ws;

    // f32 region
    float* buf1 = ws + 0 * E;                  // GEMM f32 out
    float* buf2 = ws + 1 * E;                  // inp_l / fused
    float* buf0 = ws + 2 * E;                  // sem chain f32
    float* feats[3] = { ws + 3 * E, ws + 4 * E, ws + 5 * E };
    float* small  = ws + 6 * E;
    float* cmean  = small;
    float* sphid  = cmean + BATCH * 3 * HSZ;
    float* rel    = sphid + BATCH * HSZ;
    float* node   = rel + 32;
    float* ehid   = node + BATCH * HSZ;
    float* edges  = ehid + BATCH * 3 * HSZ;
    float* graph  = edges + 32;

    // f16 region
    half_t* hp = (half_t*)(ws + 6 * E + 32768);
    half_t* xh     = hp + 0 * E;
    half_t* h16a   = hp + 1 * E;
    half_t* h16b   = hp + 2 * E;
    half_t* fwph   = hp + 3 * E;
    half_t* Qh     = hp + 4 * E;
    half_t* Kh     = hp + 5 * E;
    half_t* Vh     = hp + 6 * E;
    half_t* ctxh   = hp + 7 * E;
    half_t* conch  = hp + 8 * E;               // 3E halves
    half_t* Vt     = hp + 11 * E;              // B*NHEAD*HDIM*SEQP halves
    half_t* wgt    = Vt + (size_t)BATCH * NHEAD * HDIM * SEQP;
    half_t* semWh = wgt;            half_t* Wqh = wgt + 3 * HH;
    half_t* Wkh   = wgt + 6 * HH;   half_t* Wvh = wgt + 9 * HH;
    half_t* Woh   = wgt + 12 * HH;  half_t* fusWh = wgt + 15 * HH;

    const dim3 gG((M + 127) / 128, HSZ / 64);   // (32, 12)
    const dim3 gE(M * HSZ / 256);               // elementwise, exact
    const int gW = (int)(3 * HH / 4 / 256);     // weight cvt blocks (exact: 1728)

    // --- one-time f32 -> f16 conversions (weights + input) ---
    cvt_f16_kernel<<<gW, 256, 0, stream>>>(sem_W, semWh, (int)(3 * HH / 4));
    cvt_f16_kernel<<<gW, 256, 0, stream>>>(Wq,    Wqh,   (int)(3 * HH / 4));
    cvt_f16_kernel<<<gW, 256, 0, stream>>>(Wk,    Wkh,   (int)(3 * HH / 4));
    cvt_f16_kernel<<<gW, 256, 0, stream>>>(Wv,    Wvh,   (int)(3 * HH / 4));
    cvt_f16_kernel<<<gW, 256, 0, stream>>>(Wo,    Woh,   (int)(3 * HH / 4));
    cvt_f16_kernel<<<gW, 256, 0, stream>>>(fus_W, fusWh, (int)(3 * HH / 4));
    cvt_f16_kernel<<<(int)(E / 4 / 256), 256, 0, stream>>>(x, xh, (int)(E / 4));

    // --- semantic encoder: 3 residual Linear+LN+ReLU blocks ---
    gemm_f16wmma_kernel<<<gG, 256, 0, stream>>>(xh,   semWh,          sem_b,           nullptr, buf1, nullptr, 1.f, M, HSZ, HSZ);
    ln_act_kernel<<<M, 256, 0, stream>>>(buf1, sem_g,           sem_be,           x,    buf0, h16a, 1);
    gemm_f16wmma_kernel<<<gG, 256, 0, stream>>>(h16a, semWh + HH,     sem_b + HSZ,     nullptr, buf1, nullptr, 1.f, M, HSZ, HSZ);
    ln_act_kernel<<<M, 256, 0, stream>>>(buf1, sem_g + HSZ,     sem_be + HSZ,     buf0, buf2, h16b, 1);
    gemm_f16wmma_kernel<<<gG, 256, 0, stream>>>(h16b, semWh + 2 * HH, sem_b + 2 * HSZ, nullptr, buf1, nullptr, 1.f, M, HSZ, HSZ);
    ln_act_kernel<<<M, 256, 0, stream>>>(buf1, sem_g + 2 * HSZ, sem_be + 2 * HSZ, buf2, buf0, nullptr, 1);

    // --- hierarchical attention (Q pre-scaled by 1/sqrt(HD) in epilogue) ---
    const float* cur = buf0;
    for (int l = 0; l < NLVL; ++l) {
        add_pos_kernel<<<gE, 256, 0, stream>>>(cur, level_emb + l * HSZ,
                                               pos_emb + (size_t)l * SEQ * HSZ,
                                               1.0f + 0.1f * l, buf2, fwph);
        gemm_f16wmma_kernel<<<gG, 256, 0, stream>>>(fwph, Wqh + l * HH, bq + l * HSZ, nullptr, nullptr, Qh, 0.125f, M, HSZ, HSZ);
        gemm_f16wmma_kernel<<<gG, 256, 0, stream>>>(fwph, Wkh + l * HH, bk + l * HSZ, nullptr, nullptr, Kh, 1.0f,   M, HSZ, HSZ);
        gemm_f16wmma_kernel<<<gG, 256, 0, stream>>>(fwph, Wvh + l * HH, bv + l * HSZ, nullptr, nullptr, Vh, 1.0f,   M, HSZ, HSZ);
        if (l < 2) {
            band_attn_kernel<<<dim3(SEQ, NHEAD, BATCH), 32, 0, stream>>>(Qh, Kh, Vh, ctxh, l == 0 ? 2 : 5);
        } else {
            transposeV_kernel<<<(BATCH * NHEAD * HDIM * SEQP) / 256, 256, 0, stream>>>(Vh, Vt);
            flash_attn_kernel<<<dim3((SEQ + 15) / 16, NHEAD, BATCH), 32, 0, stream>>>(Qh, Kh, Vt, ctxh);
        }
        gemm_f16wmma_kernel<<<gG, 256, 0, stream>>>(ctxh, Woh + l * HH, bo + l * HSZ, buf2, buf1, nullptr, 1.f, M, HSZ, HSZ);
        ln_act_kernel<<<M, 256, 0, stream>>>(buf1, ln_g + l * HSZ, ln_b + l * HSZ, nullptr, feats[l], nullptr, 0);
        cur = feats[l];
    }

    // --- spurious correlation gates ---
    for (int l = 0; l < NLVL; ++l)
        meanS_kernel<<<BATCH * HSZ / 256, 256, 0, stream>>>(feats[l], cmean + l * HSZ);
    sp1_kernel<<<(BATCH * HSZ + 255) / 256, 256, 0, stream>>>(cmean, sp_W1, sp_b1, sphid);
    sp2_kernel<<<1, 32, 0, stream>>>(sphid, sp_W2, sp_b2, rel);
    for (int l = 0; l < NLVL; ++l)
        gate_concat_kernel<<<gE, 256, 0, stream>>>(feats[l], rel, conch, l);

    // --- attribute graph ---
    node_kernel<<<(BATCH * HSZ + 255) / 256, 256, 0, stream>>>(cmean, rel, node_W, node_b, node);
    edge1_kernel<<<(BATCH * 3 * HSZ + 255) / 256, 256, 0, stream>>>(node, attr_emb, edge_W1, edge_b1, ehid);
    edge2_kernel<<<1, 32, 0, stream>>>(ehid, edge_W2, edge_b2, edges);
    graph_kernel<<<(BATCH * HSZ + 255) / 256, 256, 0, stream>>>(node, attr_emb, edges, graph);

    // --- fusion + broadcast graph ---
    gemm_f16wmma_kernel<<<gG, 256, 0, stream>>>(conch, fusWh, fus_b, nullptr, buf1, nullptr, 1.f, M, HSZ, 3 * HSZ);
    ln_act_kernel<<<M, 256, 0, stream>>>(buf1, fus_g, fus_be, nullptr, buf2, nullptr, 1);
    final_add_kernel<<<gE, 256, 0, stream>>>(buf2, graph, (float*)d_out);
}
